// MultiHeadAttentionBlock_22746146800125
// MI455X (gfx1250) — matrix operand order repaired and checked by reading, not compiled, not run on hardware
//
#include <hip/hip_runtime.h>
#include <hip/hip_bf16.h>
#include <math.h>

#define DM   512
#define HEADS 8
#define DK   64
#define SEQ  4096
#define BATCH 2
#define ROWS (BATCH * SEQ) // 8192

typedef __attribute__((ext_vector_type(16))) __bf16 bf16x16;
typedef __attribute__((ext_vector_type(8)))  float  f32x8;
typedef __attribute__((ext_vector_type(4)))  float  f32x4;
typedef __attribute__((ext_vector_type(4)))  unsigned int u32x4;

union BFrag  { bf16x16 v; u32x4 q[2]; };
union BPack8 { u32x4 q; __bf16 h[8]; };

// ---- WMMA fragment builders (CDNA5 16x16x32 bf16 layouts, wave32) ----
// A (16x32, MxK): lane l holds row m=l&15; elems 0..7 = K[(l>>4)*8 + 0..7],
// elems 8..15 = K[16 + (l>>4)*8 + 0..7]  -> two contiguous 16B loads.
__device__ __forceinline__ bf16x16 frag_a(const __bf16* A, int ld, int m0, int k0) {
  const int lane = threadIdx.x & 31;
  const __bf16* p = A + (size_t)(m0 + (lane & 15)) * ld + k0 + ((lane >> 4) << 3);
  BFrag f;
  f.q[0] = *(const u32x4*)p;
  f.q[1] = *(const u32x4*)(p + 16);
  return f.v;
}
// B (32x16, KxN) supplied as B^T row-major (ld = K-stride of B^T rows):
// lane l holds col n=l&15; elems 0..7 = K[(l>>4)*8 + 0..7],
// elems 8..15 = K[16 + (l>>4)*8 + 0..7] -> two contiguous 16B loads.
__device__ __forceinline__ bf16x16 frag_b(const __bf16* Bt, int ld, int n0, int k0) {
  const int lane = threadIdx.x & 31;
  const __bf16* p = Bt + (size_t)(n0 + (lane & 15)) * ld + k0 + ((lane >> 4) << 3);
  BFrag f;
  f.q[0] = *(const u32x4*)p;
  f.q[1] = *(const u32x4*)(p + 16);
  return f.v;
}
// C/D (16x16 f32): lane l holds col n=l&15, elem j = row (l>>4)*8 + j.
__device__ __forceinline__ f32x8 wmma_bf16(bf16x16 a, bf16x16 b, f32x8 c) {
  return __builtin_amdgcn_wmma_f32_16x16x32_bf16(false, a, false, b, (short)0, c,
                                                 false, false);
}

// ---- Kernel 0: fp32 -> bf16 weight conversion ----
__global__ void k_f32_to_bf16(const float* __restrict__ src,
                              __bf16* __restrict__ dst, int n) {
  int i = blockIdx.x * blockDim.x + threadIdx.x;
  if (i < n) dst[i] = (__bf16)src[i];
}

// ---- Kernel 1: fused QKV projection ----
// z=0: Qh[b,h,s,d]=q@Wq^T ; z=1: Kh ; z=2: Vt[b,h,d,s] (transposed for PV B-frags)
__global__ __launch_bounds__(256)
void k_qkv_proj(const float* __restrict__ Xq, const float* __restrict__ Xk,
                const float* __restrict__ Xv,
                const __bf16* __restrict__ Wq, const __bf16* __restrict__ Wk,
                const __bf16* __restrict__ Wv,
                __bf16* __restrict__ Qh, __bf16* __restrict__ Kh,
                __bf16* __restrict__ Vt) {
  __shared__ __bf16 sX[128 * 40]; // 128 rows x 32 k, pad to 40 (16B-aligned rows)
  const int z = blockIdx.z;
  const float*  X   = (z == 0) ? Xq : (z == 1) ? Xk : Xv;
  const __bf16* W   = (z == 0) ? Wq : (z == 1) ? Wk : Wv;
  __bf16*       Out = (z == 0) ? Qh : (z == 1) ? Kh : Vt;
  const int head  = blockIdx.y;
  const int rows0 = blockIdx.x * 128;
  const int wave  = threadIdx.x >> 5;
  const int lane  = threadIdx.x & 31;
  const int n15 = lane & 15, h16 = lane >> 4;

  f32x8 acc[4] = {};
  const int lr = threadIdx.x >> 1;        // 0..127
  const int lc = (threadIdx.x & 1) * 16;  // 0 / 16

  for (int kc = 0; kc < DM; kc += 32) {
    __syncthreads();
    { // stage 128x32 fp32 tile -> bf16 LDS
      const float* src = X + (size_t)(rows0 + lr) * DM + kc + lc;
      f32x4 f0 = *(const f32x4*)(src + 0);
      f32x4 f1 = *(const f32x4*)(src + 4);
      f32x4 f2 = *(const f32x4*)(src + 8);
      f32x4 f3 = *(const f32x4*)(src + 12);
      BPack8 p0, p1;
#pragma unroll
      for (int i = 0; i < 4; ++i) {
        p0.h[i] = (__bf16)f0[i]; p0.h[4 + i] = (__bf16)f1[i];
        p1.h[i] = (__bf16)f2[i]; p1.h[4 + i] = (__bf16)f3[i];
      }
      *(u32x4*)&sX[lr * 40 + lc]     = p0.q;
      *(u32x4*)&sX[lr * 40 + lc + 8] = p1.q;
    }
    __syncthreads();
    bf16x16 a = frag_a(sX, 40, wave * 16, 0);
#pragma unroll
    for (int nt = 0; nt < 4; ++nt) { // W rows == B^T rows
      bf16x16 b = frag_b(W, DM, head * 64 + nt * 16, kc);
      acc[nt] = wmma_bf16(a, b, acc[nt]);
    }
  }

  const int grb = rows0 + wave * 16 + h16 * 8; // global row for j=0
  const int bb = grb >> 12, ss = grb & (SEQ - 1);
  if (z < 2) {
#pragma unroll
    for (int nt = 0; nt < 4; ++nt) {
      const int d = nt * 16 + n15;
#pragma unroll
      for (int j = 0; j < 8; ++j)
        Out[(((size_t)bb * HEADS + head) * SEQ + (ss + j)) * DK + d] =
            (__bf16)acc[nt][j];
    }
  } else { // V transposed: 8 consecutive s per lane -> 16B store
#pragma unroll
    for (int nt = 0; nt < 4; ++nt) {
      const int d = nt * 16 + n15;
      BPack8 p;
#pragma unroll
      for (int j = 0; j < 8; ++j) p.h[j] = (__bf16)acc[nt][j];
      *(u32x4*)(Out + (((size_t)bb * HEADS + head) * DK + d) * SEQ + ss) = p.q;
    }
  }
}

// ---- Kernel 2: flash attention (online softmax, 128 q rows/block) ----
__global__ __launch_bounds__(256)
void k_attention(const __bf16* __restrict__ Qh, const __bf16* __restrict__ Kh,
                 const __bf16* __restrict__ Vt, const int* __restrict__ Mask,
                 __bf16* __restrict__ Ctx) {
  __shared__ __bf16 sP[8 * 16 * 72]; // per-wave 16x64 P tile (pad 72)
  const int bh = blockIdx.y, b = bh >> 3, h = bh & 7;
  const int wave = threadIdx.x >> 5, lane = threadIdx.x & 31;
  const int n15 = lane & 15, h16 = lane >> 4;
  const int m0 = blockIdx.x * 128 + wave * 16;

  const __bf16* Qb = Qh + (size_t)bh * SEQ * DK;
  const __bf16* Kb = Kh + (size_t)bh * SEQ * DK;
  const __bf16* Vb = Vt + (size_t)bh * DK * SEQ;
  const int* Mb = Mask + (size_t)b * SEQ;
  __bf16* sPw = sP + wave * 16 * 72;

  bf16x16 aQ0 = frag_a(Qb, DK, m0, 0);
  bf16x16 aQ1 = frag_a(Qb, DK, m0, 32);

  f32x8 o[4] = {};
  float mi[8], li[8];
#pragma unroll
  for (int j = 0; j < 8; ++j) { mi[j] = -1e30f; li[j] = 0.f; }

  for (int kv = 0; kv < SEQ; kv += 64) {
    if (kv + 64 < SEQ) {
      __builtin_prefetch(Kb + (size_t)(kv + 64) * DK + lane * 8, 0, 3);
      __builtin_prefetch(Vb + (size_t)n15 * SEQ + kv + 64, 0, 3);
    }
    // S = Q K^T (K rows are B^T rows), scale + mask
    f32x8 sc[4];
#pragma unroll
    for (int nt = 0; nt < 4; ++nt) {
      bf16x16 b0 = frag_b(Kb, DK, kv + nt * 16, 0);
      bf16x16 b1 = frag_b(Kb, DK, kv + nt * 16, 32);
      f32x8 s = {};
      s = wmma_bf16(aQ0, b0, s);
      s = wmma_bf16(aQ1, b1, s);
      const int mv = Mb[kv + nt * 16 + n15];
#pragma unroll
      for (int j = 0; j < 8; ++j)
        sc[nt][j] = mv ? s[j] * 0.125f : -1.0e9f;
    }
    // row max: rows live in 16-lane groups (C layout) -> shfl_xor width 16
    float corr[8];
#pragma unroll
    for (int j = 0; j < 8; ++j) {
      float v = fmaxf(fmaxf(sc[0][j], sc[1][j]), fmaxf(sc[2][j], sc[3][j]));
#pragma unroll
      for (int off = 8; off >= 1; off >>= 1)
        v = fmaxf(v, __shfl_xor(v, off, 16));
      float mn = fmaxf(mi[j], v);
      corr[j] = __expf(mi[j] - mn);
      mi[j] = mn;
    }
    // P = exp(S - m), row sums, repack P -> LDS as bf16 A tile
    float rsum[8];
#pragma unroll
    for (int j = 0; j < 8; ++j) rsum[j] = 0.f;
#pragma unroll
    for (int nt = 0; nt < 4; ++nt)
#pragma unroll
      for (int j = 0; j < 8; ++j) {
        float p = __expf(sc[nt][j] - mi[j]);
        rsum[j] += p;
        sPw[(h16 * 8 + j) * 72 + nt * 16 + n15] = (__bf16)p;
      }
#pragma unroll
    for (int j = 0; j < 8; ++j) {
      float v = rsum[j];
#pragma unroll
      for (int off = 8; off >= 1; off >>= 1)
        v += __shfl_xor(v, off, 16);
      li[j] = li[j] * corr[j] + v;
    }
#pragma unroll
    for (int nt = 0; nt < 4; ++nt)
#pragma unroll
      for (int j = 0; j < 8; ++j) o[nt][j] *= corr[j];
    // O += P V  (Vt rows are B^T rows; same-wave LDS RAW is in-order)
    bf16x16 aP0 = frag_a(sPw, 72, 0, 0);
    bf16x16 aP1 = frag_a(sPw, 72, 0, 32);
#pragma unroll
    for (int nt = 0; nt < 4; ++nt) {
      bf16x16 v0 = frag_b(Vb, SEQ, nt * 16, kv);
      bf16x16 v1 = frag_b(Vb, SEQ, nt * 16, kv + 32);
      o[nt] = wmma_bf16(aP0, v0, o[nt]);
      o[nt] = wmma_bf16(aP1, v1, o[nt]);
    }
  }
  // normalize, store context [B,S,512] bf16
  const int ss = m0 + h16 * 8;
#pragma unroll
  for (int nt = 0; nt < 4; ++nt) {
    const int col = h * DK + nt * 16 + n15;
#pragma unroll
    for (int j = 0; j < 8; ++j)
      Ctx[((size_t)b * SEQ + ss + j) * DM + col] = (__bf16)(o[nt][j] / li[j]);
  }
}

// ---- Kernel 3: output projection, fp32 result ----
__global__ __launch_bounds__(256)
void k_out_proj(const __bf16* __restrict__ Ctx, const __bf16* __restrict__ Wo,
                float* __restrict__ Out) {
  const int n0 = blockIdx.y * 64;
  const int wave = threadIdx.x >> 5, lane = threadIdx.x & 31;
  const int n15 = lane & 15, h16 = lane >> 4;
  const int m0 = blockIdx.x * 128 + wave * 16;

  f32x8 acc[4] = {};
  for (int kc = 0; kc < DM; kc += 32) {
    bf16x16 a = frag_a(Ctx, DM, m0, kc);
#pragma unroll
    for (int nt = 0; nt < 4; ++nt) {
      bf16x16 bfr = frag_b(Wo, DM, n0 + nt * 16, kc);
      acc[nt] = wmma_bf16(a, bfr, acc[nt]);
    }
  }
  const int grb = m0 + h16 * 8;
#pragma unroll
  for (int nt = 0; nt < 4; ++nt) {
    const int c = n0 + nt * 16 + n15;
#pragma unroll
    for (int j = 0; j < 8; ++j)
      Out[(size_t)(grb + j) * DM + c] = acc[nt][j];
  }
}

extern "C" void kernel_launch(void* const* d_in, const int* in_sizes, int n_in,
                              void* d_out, int out_size, void* d_ws, size_t ws_size,
                              hipStream_t stream) {
  const float* q    = (const float*)d_in[0];
  const float* k    = (const float*)d_in[1];
  const float* v    = (const float*)d_in[2];
  const int*   mask = (const int*)d_in[3];
  const float* wq   = (const float*)d_in[4];
  const float* wk   = (const float*)d_in[5];
  const float* wv   = (const float*)d_in[6];
  const float* wo   = (const float*)d_in[7];

  char* ws = (char*)d_ws;
  const size_t WSZ = (size_t)DM * DM * 2;                 // 512 KB (bf16 weight)
  const size_t HSZ = (size_t)BATCH * HEADS * SEQ * DK * 2; // 8 MB per tensor
  __bf16* wqb = (__bf16*)(ws);
  __bf16* wkb = (__bf16*)(ws + WSZ);
  __bf16* wvb = (__bf16*)(ws + 2 * WSZ);
  __bf16* wob = (__bf16*)(ws + 3 * WSZ);
  __bf16* Qh  = (__bf16*)(ws + 4 * WSZ);
  __bf16* Kh  = (__bf16*)(ws + 4 * WSZ + HSZ);
  __bf16* Vt  = (__bf16*)(ws + 4 * WSZ + 2 * HSZ);
  __bf16* Ctx = (__bf16*)(ws + 4 * WSZ + 3 * HSZ);
  (void)in_sizes; (void)n_in; (void)out_size; (void)ws_size;

  const int wn = DM * DM;
  k_f32_to_bf16<<<wn / 256, 256, 0, stream>>>(wq, wqb, wn);
  k_f32_to_bf16<<<wn / 256, 256, 0, stream>>>(wk, wkb, wn);
  k_f32_to_bf16<<<wn / 256, 256, 0, stream>>>(wv, wvb, wn);
  k_f32_to_bf16<<<wn / 256, 256, 0, stream>>>(wo, wob, wn);

  k_qkv_proj<<<dim3(ROWS / 128, HEADS, 3), 256, 0, stream>>>(
      q, k, v, wqb, wkb, wvb, Qh, Kh, Vt);
  k_attention<<<dim3(SEQ / 128, BATCH * HEADS), 256, 0, stream>>>(
      Qh, Kh, Vt, mask, Ctx);
  k_out_proj<<<dim3(ROWS / 128, DM / 64), 256, 0, stream>>>(
      Ctx, wob, (float*)d_out);
}
